// ParallelCoattentionModel_80839874445840
// MI455X (gfx1250) — matrix operand order repaired and checked by reading, not compiled, not run on hardware
//
#include <hip/hip_runtime.h>
#include <math.h>

// ---------------- dims (hardcoded from reference) ----------------
#define DD   512
#define BB   256
#define RR   196
#define TT   26
#define AV   3000

// ---------------- WMMA types ----------------
typedef __attribute__((ext_vector_type(16))) __bf16 v16bf;
typedef __attribute__((ext_vector_type(2)))  __bf16 v2bf;
typedef __attribute__((ext_vector_type(8)))  float  v8f;
typedef unsigned long long u64t;

union FragBF {
  v16bf v;
  u64t  q[4];
};
union Chunk8 {
  unsigned w[4];
  u64t     q[2];
};

// pack two f32 -> two bf16 (RNE) in one dword; standard fptrunc IR so the
// backend can select the native pack-convert
__device__ __forceinline__ unsigned cvt2_bf16(float lo, float hi) {
  v2bf r = { (__bf16)lo, (__bf16)hi };
  return __builtin_bit_cast(unsigned, r);
}

__device__ __forceinline__ float fast_tanh(float x) {
  float e = __expf(-2.0f * x);
  return (1.0f - e) / (1.0f + e);
}
__device__ __forceinline__ float fast_sigmoid(float x) {
  return 1.0f / (1.0f + __expf(-x));
}

// =================================================================
// Tiled bf16 WMMA GEMM:  C[M,N] = epi( A[M,K] @ B[K,N] )
//   block tile 128x64, 8 waves (4x2), each wave: 2x2 16x16 WMMA tiles
//   LDS staged in fragment order (per-lane 40B slots -> ds_load_b64 x4)
// Template params:
//   CONV    : A element (row r=b*T+t, k) = word[b, t*D + koff + k], zero-padded
//             (chunk-granular pad test: 8-aligned base vs 8-aligned signal len)
//   OUTMODE : 0 store, 1 max-with-existing C
//   ACT     : 0 none, 1 tanh
//   HASADD  : epilogue adds addend[row*ldadd + col]
//   BOUNDS  : ragged N only (M,K exact for every call site)
// =================================================================
#define BM 128
#define BN 64
#define BKK 32

template <int CONV, int OUTMODE, int ACT, int HASADD, int BOUNDS>
__global__ __launch_bounds__(256)
void wmma_gemm(const float* __restrict__ A, int lda,
               const float* __restrict__ Bm, int ldb,
               const float* __restrict__ bias, float bias_scale,
               const float* __restrict__ addend, int ldadd,
               float* __restrict__ C, int ldc,
               int M, int N, int K, int conv_koff) {
  // fragment-order tiles: [16-row/col tile][lane][20 ushort slot (16 used + 4 pad)]
  __shared__ unsigned short lA[8][32][20];   // 128 x 32  (10.0 KB)
  __shared__ unsigned short lB[4][32][20];   //  32 x 64  ( 5.0 KB)

  const int tid  = threadIdx.x;
  const int lane = tid & 31;
  const int w    = tid >> 5;
  const int wm   = w >> 1;          // 0..3 : 32-row strip
  const int wn   = w & 1;           // 0..1 : 32-col strip
  const int blockM = blockIdx.y * BM;
  const int blockN = blockIdx.x * BN;

  // ---- K-invariant staging setup ----
  // A: two chunks of 8 consecutive k per thread
  u64t*        a_dst[2];
  const float* a_src[2];
  int          a_tb[2];
  #pragma unroll
  for (int half = 0; half < 2; ++half) {
    const int cid = tid + (half << 8);
    const int m   = cid >> 2;                 // 0..127
    const int kc  = (cid & 3) << 3;           // 0,8,16,24
    a_dst[half] = (u64t*)&lA[m >> 4][(m & 15) + (((kc >> 3) & 1) << 4)][(kc >> 4) << 3];
    const int gm = blockM + m;
    if (CONV) {
      const int b = gm / TT, t = gm - b * TT;
      a_src[half] = A + (long)b * (TT * DD);
      a_tb[half]  = t * DD + conv_koff + kc;  // 8-aligned chunk base (k0 added later)
    } else {
      a_src[half] = A + (long)gm * lda + kc;
      a_tb[half]  = 0;
    }
  }
  // B: one chunk of 8 consecutive k at fixed column
  const int sb_n  = tid >> 2;            // 0..63
  const int sb_kc = (tid & 3) << 3;      // 0,8,16,24
  const int gn    = blockN + sb_n;
  u64t* b_dst = (u64t*)&lB[sb_n >> 4][(sb_n & 15) + (((sb_kc >> 3) & 1) << 4)][(sb_kc >> 4) << 3];
  const float* b_src = Bm + (long)sb_kc * ldb + gn;
  const bool b_ok = (!BOUNDS) || (gn < N);

  v8f acc00 = {}, acc01 = {}, acc10 = {}, acc11 = {};

  for (int k0 = 0; k0 < K; k0 += BKK) {
    // ---------- stage A ----------
    #pragma unroll
    for (int half = 0; half < 2; ++half) {
      Chunk8 t8;
      if (CONV) {
        const int base = a_tb[half] + k0;
        if ((unsigned)base <= (unsigned)(TT * DD - 8)) {   // whole chunk in-range
          const float4* p = (const float4*)(a_src[half] + base);
          float4 v0 = p[0], v1 = p[1];
          t8.w[0] = cvt2_bf16(v0.x, v0.y); t8.w[1] = cvt2_bf16(v0.z, v0.w);
          t8.w[2] = cvt2_bf16(v1.x, v1.y); t8.w[3] = cvt2_bf16(v1.z, v1.w);
        } else {
          t8.q[0] = 0; t8.q[1] = 0;                        // zero padding
        }
      } else {
        const float4* p = (const float4*)(a_src[half] + k0);
        float4 v0 = p[0], v1 = p[1];
        t8.w[0] = cvt2_bf16(v0.x, v0.y); t8.w[1] = cvt2_bf16(v0.z, v0.w);
        t8.w[2] = cvt2_bf16(v1.x, v1.y); t8.w[3] = cvt2_bf16(v1.z, v1.w);
      }
      a_dst[half][0] = t8.q[0];
      a_dst[half][1] = t8.q[1];
    }
    // ---------- stage B ----------
    {
      Chunk8 t8;
      if (b_ok) {
        const float* p = b_src + (long)k0 * ldb;
        float f0 = p[0];            float f1 = p[(long)ldb];
        float f2 = p[2L * ldb];     float f3 = p[3L * ldb];
        float f4 = p[4L * ldb];     float f5 = p[5L * ldb];
        float f6 = p[6L * ldb];     float f7 = p[7L * ldb];
        t8.w[0] = cvt2_bf16(f0, f1); t8.w[1] = cvt2_bf16(f2, f3);
        t8.w[2] = cvt2_bf16(f4, f5); t8.w[3] = cvt2_bf16(f6, f7);
      } else {
        t8.q[0] = 0; t8.q[1] = 0;
      }
      b_dst[0] = t8.q[0];
      b_dst[1] = t8.q[1];
    }
    __syncthreads();

    // ---------- fragments: contiguous 32B per lane -> ds_load_b64 x4 ----------
    FragBF a0, a1, b0, b1;
    {
      const u64t* pa0 = (const u64t*)&lA[(wm << 1) + 0][lane][0];
      const u64t* pa1 = (const u64t*)&lA[(wm << 1) + 1][lane][0];
      const u64t* pb0 = (const u64t*)&lB[(wn << 1) + 0][lane][0];
      const u64t* pb1 = (const u64t*)&lB[(wn << 1) + 1][lane][0];
      #pragma unroll
      for (int i = 0; i < 4; ++i) {
        a0.q[i] = pa0[i]; a1.q[i] = pa1[i];
        b0.q[i] = pb0[i]; b1.q[i] = pb1[i];
      }
    }
    acc00 = __builtin_amdgcn_wmma_f32_16x16x32_bf16(false, a0.v, false, b0.v, (short)0, acc00, false, false);
    acc01 = __builtin_amdgcn_wmma_f32_16x16x32_bf16(false, a0.v, false, b1.v, (short)0, acc01, false, false);
    acc10 = __builtin_amdgcn_wmma_f32_16x16x32_bf16(false, a1.v, false, b0.v, (short)0, acc10, false, false);
    acc11 = __builtin_amdgcn_wmma_f32_16x16x32_bf16(false, a1.v, false, b1.v, (short)0, acc11, false, false);
    __syncthreads();
  }

  // ---------------- epilogue ----------------
  #pragma unroll
  for (int i = 0; i < 2; ++i) {
    #pragma unroll
    for (int j = 0; j < 2; ++j) {
      const v8f* accp = (i == 0) ? (j == 0 ? &acc00 : &acc01)
                                 : (j == 0 ? &acc10 : &acc11);
      const int gcol = blockN + (wn << 5) + (j << 4) + (lane & 15);
      const int rowb = blockM + (wm << 5) + (i << 4) + ((lane >> 4) << 3);
      if (!BOUNDS || gcol < N) {
        const float bv = bias ? bias[gcol] * bias_scale : 0.0f;
        #pragma unroll
        for (int v = 0; v < 8; ++v) {
          const int grow = rowb + v;
          float r = (*accp)[v] + bv;
          if (HASADD) r += addend[(long)grow * ldadd + gcol];
          if (ACT == 1) r = fast_tanh(r);
          long off = (long)grow * ldc + gcol;
          if (OUTMODE == 1) { float e = C[off]; C[off] = fmaxf(e, r); }
          else              { C[off] = r; }
        }
      }
    }
  }
}

// ---------------- elementwise / reduction kernels ----------------
__global__ void zero_kernel(float* p, int n) {
  int i = blockIdx.x * blockDim.x + threadIdx.x;
  if (i < n) p[i] = 0.0f;
}

__global__ void gather_embed_kernel(const int* __restrict__ q,
                                    const float* __restrict__ emb,
                                    float* __restrict__ word, int total) {
  int i = blockIdx.x * blockDim.x + threadIdx.x;
  if (i < total) {
    int row = i >> 9;
    int d   = i & 511;
    word[i] = emb[(long)q[row] * DD + d];
  }
}

// out[b,d] = sum_n in[b,n,d]
__global__ void colsum_kernel(const float* __restrict__ in, float* __restrict__ out,
                              int Bn, int Nmid) {
  int i = blockIdx.x * blockDim.x + threadIdx.x;
  if (i < Bn * DD) {
    int b = i >> 9, d = i & 511;
    const float* p = in + (long)b * Nmid * DD + d;
    float s = 0.0f;
    for (int n = 0; n < Nmid; ++n) s += p[(long)n * DD];
    out[i] = s;
  }
}

// Keras LSTM gates (i,f,c,o); accumulates sum-over-time of h into ss
__global__ void lstm_gate_kernel(const float* __restrict__ g,
                                 float* __restrict__ h, float* __restrict__ c,
                                 float* __restrict__ ss, int n) {
  int i = blockIdx.x * blockDim.x + threadIdx.x;
  if (i < n) {
    int b = i >> 9, d = i & 511;
    const float* gr = g + (long)b * (4 * DD);
    float ig = gr[d], fg = gr[DD + d], cg = gr[2 * DD + d], og = gr[3 * DD + d];
    float cc = fast_sigmoid(fg) * c[i] + fast_sigmoid(ig) * fast_tanh(cg);
    float hh = fast_sigmoid(og) * fast_tanh(cc);
    c[i] = cc; h[i] = hh; ss[i] += hh;
  }
}

__global__ void add2_kernel(float* __restrict__ dst, const float* __restrict__ x,
                            const float* __restrict__ y, int n) {
  int i = blockIdx.x * blockDim.x + threadIdx.x;
  if (i < n) dst[i] = x[i] + y[i];
}

// dst[rows,1024]: cols 0..511 = x+y, cols 512..1023 = z
__global__ void pack_concat_kernel(const float* __restrict__ x, const float* __restrict__ y,
                                   const float* __restrict__ z, float* __restrict__ dst,
                                   int rows) {
  int i = blockIdx.x * blockDim.x + threadIdx.x;
  if (i < rows * 2 * DD) {
    int r = i >> 10, cidx = i & 1023;
    dst[i] = (cidx < DD) ? (x[r * DD + cidx] + y[r * DD + cidx])
                         : z[r * DD + (cidx - DD)];
  }
}

// ---------------- host side ----------------
static inline dim3 ggrid(int M, int N) { return dim3((N + BN - 1) / BN, (M + BM - 1) / BM); }
static inline dim3 g1(int n) { return dim3((n + 255) / 256); }

extern "C" void kernel_launch(void* const* d_in, const int* in_sizes, int n_in,
                              void* d_out, int out_size, void* d_ws, size_t ws_size,
                              hipStream_t stream) {
  (void)in_sizes; (void)n_in; (void)out_size; (void)ws_size;
  const float* image_feat = (const float*)d_in[0];
  const int*   qenc       = (const int*)  d_in[1];
  const float* W_ip = (const float*)d_in[2];  const float* b_ip = (const float*)d_in[3];
  const float* emb  = (const float*)d_in[4];
  const float* W_uni= (const float*)d_in[5];  const float* b_uni= (const float*)d_in[6];
  const float* W_bi = (const float*)d_in[7];  const float* b_bi = (const float*)d_in[8];
  const float* W_tri= (const float*)d_in[9];  const float* b_tri= (const float*)d_in[10];
  const float* Wx   = (const float*)d_in[11]; const float* Wh   = (const float*)d_in[12];
  const float* b_lstm = (const float*)d_in[13];
  // d_in[14..23] (W_c..b_at) are provably dead: softmax over a length-1 axis == 1.0,
  // so the attention pooling reduces to plain sums independent of those weights.
  const float* W_w = (const float*)d_in[24]; const float* b_w = (const float*)d_in[25];
  const float* W_p = (const float*)d_in[26]; const float* b_p = (const float*)d_in[27];
  const float* W_s = (const float*)d_in[28]; const float* b_s = (const float*)d_in[29];
  const float* W_f = (const float*)d_in[30]; const float* b_f = (const float*)d_in[31];
  float* out = (float*)d_out;

  // ---- workspace layout (floats) ----
  float* ws = (float*)d_ws;
  const long nWord = (long)BB * TT * DD;
  const long nXg   = (long)BB * TT * 4 * DD;
  const long nG    = (long)BB * 4 * DD;
  const long nBD   = (long)BB * DD;
  float* word   = ws;                float* phrase = word + nWord;
  float* xg     = phrase + nWord;    float* g      = xg + nXg;
  float* h      = g + nG;            float* c      = h + nBD;
  float* ss     = c + nBD;           float* s_word = ss + nBD;
  float* s_phr  = s_word + nBD;      float* s_imgr = s_phr + nBD;
  float* s_img  = s_imgr + nBD;      float* u_w    = s_img + nBD;
  float* h_w    = u_w + nBD;         float* catP   = h_w + nBD;
  float* h_p    = catP + 2 * nBD;    float* catS   = h_p + nBD;
  float* h_s    = catS + 2 * nBD;

  const int MT = BB * TT;  // 6656

  // 0) zero LSTM state (h,c,ss contiguous)
  zero_kernel<<<g1((int)(3 * nBD)), 256, 0, stream>>>(h, (int)(3 * nBD));

  // 1) word = emb[q]; s_word = sum_t word
  gather_embed_kernel<<<g1((int)nWord), 256, 0, stream>>>(qenc, emb, word, (int)nWord);
  colsum_kernel<<<g1((int)nBD), 256, 0, stream>>>(word, s_word, BB, TT);

  // 2) phrase = max(uni, bi, tri): strided convs as shifted-window GEMMs, fused max
  wmma_gemm<1,0,0,0,0><<<ggrid(MT, DD), 256, 0, stream>>>(
      word, 0, W_uni, DD, b_uni, 1.0f, nullptr, 0, phrase, DD, MT, DD, DD, 0);
  wmma_gemm<1,1,0,0,0><<<ggrid(MT, DD), 256, 0, stream>>>(
      word, 0, W_bi, DD, b_bi, 1.0f, nullptr, 0, phrase, DD, MT, DD, 2 * DD, -256);
  wmma_gemm<1,1,0,0,0><<<ggrid(MT, DD), 256, 0, stream>>>(
      word, 0, W_tri, DD, b_tri, 1.0f, nullptr, 0, phrase, DD, MT, DD, 3 * DD, -512);
  colsum_kernel<<<g1((int)nBD), 256, 0, stream>>>(phrase, s_phr, BB, TT);

  // 3) LSTM: xg = phrase @ Wx + b_lstm; 26 recurrent steps, ss += h each step
  wmma_gemm<0,0,0,0,0><<<ggrid(MT, 4 * DD), 256, 0, stream>>>(
      phrase, DD, Wx, 4 * DD, b_lstm, 1.0f, nullptr, 0, xg, 4 * DD, MT, 4 * DD, DD, 0);
  for (int t = 0; t < TT; ++t) {
    wmma_gemm<0,0,0,1,0><<<ggrid(BB, 4 * DD), 256, 0, stream>>>(
        h, DD, Wh, 4 * DD, nullptr, 0.0f, xg + (long)t * 4 * DD, TT * 4 * DD,
        g, 4 * DD, BB, 4 * DD, DD, 0);
    lstm_gate_kernel<<<g1((int)nBD), 256, 0, stream>>>(g, h, c, ss, (int)nBD);
  }

  // 4) s_img = (sum_r image_feat) @ W_ip + R*b_ip
  colsum_kernel<<<g1((int)nBD), 256, 0, stream>>>(image_feat, s_imgr, BB, RR);
  wmma_gemm<0,0,0,0,0><<<ggrid(BB, DD), 256, 0, stream>>>(
      s_imgr, DD, W_ip, DD, b_ip, (float)RR, nullptr, 0, s_img, DD, BB, DD, DD, 0);

  // 5) head: h_w -> h_p -> h_s -> logits
  add2_kernel<<<g1((int)nBD), 256, 0, stream>>>(u_w, s_img, s_word, (int)nBD);
  wmma_gemm<0,0,1,0,0><<<ggrid(BB, DD), 256, 0, stream>>>(
      u_w, DD, W_w, DD, b_w, 1.0f, nullptr, 0, h_w, DD, BB, DD, DD, 0);
  pack_concat_kernel<<<g1((int)(2 * nBD)), 256, 0, stream>>>(s_img, s_phr, h_w, catP, BB);
  wmma_gemm<0,0,1,0,0><<<ggrid(BB, DD), 256, 0, stream>>>(
      catP, 2 * DD, W_p, DD, b_p, 1.0f, nullptr, 0, h_p, DD, BB, DD, 2 * DD, 0);
  pack_concat_kernel<<<g1((int)(2 * nBD)), 256, 0, stream>>>(s_img, ss, h_p, catS, BB);
  wmma_gemm<0,0,1,0,0><<<ggrid(BB, DD), 256, 0, stream>>>(
      catS, 2 * DD, W_s, DD, b_s, 1.0f, nullptr, 0, h_s, DD, BB, DD, 2 * DD, 0);
  wmma_gemm<0,0,0,0,1><<<ggrid(BB, AV), 256, 0, stream>>>(
      h_s, DD, W_f, AV, b_f, 1.0f, nullptr, 0, out, AV, BB, AV, DD, 0);
}
